// XCAttention_28381143892840
// MI455X (gfx1250) — matrix operand order repaired and checked by reading, not compiled, not run on hardware
//
#include <hip/hip_runtime.h>

typedef __attribute__((ext_vector_type(16))) __bf16 v16bf;
typedef __attribute__((ext_vector_type(8)))  __bf16 v8bf;
typedef __attribute__((ext_vector_type(8)))  float  v8f;

union BF16Frag { v16bf v; v8bf h[2]; };

__device__ __forceinline__ unsigned short f2bf(float f) {
  unsigned u = __float_as_uint(f);
  u += 0x7FFFu + ((u >> 16) & 1u);          // round-to-nearest-even
  return (unsigned short)(u >> 16);
}
__device__ __forceinline__ float bf2f(unsigned short s) {
  return __uint_as_float(((unsigned)s) << 16);
}

// ---------------------------------------------------------------------------
// Kernel 1: FiLM MLP  f = silu(cond@W1+b1)@W2+b2   -> gb[b][1024]
// ---------------------------------------------------------------------------
__global__ void film_kernel(const float* __restrict__ cond,
                            const float* __restrict__ w1, const float* __restrict__ b1,
                            const float* __restrict__ w2, const float* __restrict__ b2,
                            float* __restrict__ gb) {
  __shared__ float hid[1024];
  const int b = blockIdx.x, tid = threadIdx.x;
  for (int jr = 0; jr < 4; ++jr) {
    int j = tid + jr * 256;
    float s = b1[j];
#pragma unroll 8
    for (int c = 0; c < 512; ++c) s += cond[b * 512 + c] * w1[c * 1024 + j];
    hid[j] = s / (1.0f + expf(-s));         // silu
  }
  __syncthreads();
  for (int jr = 0; jr < 4; ++jr) {
    int j = tid + jr * 256;
    float s = b2[j];
#pragma unroll 8
    for (int c = 0; c < 1024; ++c) s += hid[c] * w2[c * 1024 + j];
    gb[b * 1024 + j] = s;
  }
}

// ---------------------------------------------------------------------------
// Kernel 2: convert weights to bf16 (L2-resident afterwards)
// ---------------------------------------------------------------------------
__global__ void cvtw_kernel(const float* __restrict__ qw, const float* __restrict__ ow,
                            unsigned short* __restrict__ qbf, unsigned short* __restrict__ obf) {
  int i = blockIdx.x * 256 + threadIdx.x;
  if (i < 512 * 1536) qbf[i] = f2bf(qw[i]);
  int o = i - 512 * 1536;
  if (o >= 0 && o < 512 * 512) obf[o] = f2bf(ow[o]);
}

// ---------------------------------------------------------------------------
// Kernel 3: LayerNorm + FiLM, transpose (b,c,n) -> token-major bf16 t[(b,n)][c]
// ---------------------------------------------------------------------------
__global__ void ln_kernel(const float* __restrict__ x, const float* __restrict__ gb,
                          unsigned short* __restrict__ tbf) {
  __shared__ float ps1[8][32], ps2[8][32], smu[32], srs[32];
  __shared__ unsigned short ts[32 * 516];
  const int tid = threadIdx.x;
  const int g = tid >> 5, tg = tid & 31;
  const int m0 = blockIdx.x * 32;
  const int b = m0 >> 12, n0 = m0 & 4095;
  const float* xb = x + (size_t)b * 512 * 4096 + n0 + tg;
  float s1 = 0.f, s2 = 0.f;
#pragma unroll 8
  for (int i = 0; i < 64; ++i) {
    int c = g * 64 + i;
    float v = xb[(size_t)c * 4096];
    s1 += v; s2 += v * v;
  }
  ps1[g][tg] = s1; ps2[g][tg] = s2;
  __syncthreads();
  if (tid < 32) {
    float a = 0.f, q = 0.f;
#pragma unroll
    for (int gg = 0; gg < 8; ++gg) { a += ps1[gg][tid]; q += ps2[gg][tid]; }
    float mu = a * (1.0f / 512.0f);
    float var = q * (1.0f / 512.0f) - mu * mu;
    smu[tid] = mu; srs[tid] = rsqrtf(var + 1e-5f);
  }
  __syncthreads();
  float mu = smu[tg], rs = srs[tg];
#pragma unroll 8
  for (int i = 0; i < 64; ++i) {
    int c = g * 64 + i;
    float v = xb[(size_t)c * 4096];
    float ga = gb[b * 1024 + c], be = gb[b * 1024 + 512 + c];
    ts[tg * 516 + c] = f2bf((v - mu) * rs * ga + be);
  }
  __syncthreads();
  {
    ushort4 ro[16];
#pragma unroll
    for (int r = 0; r < 16; ++r) {
      int e = tid + r * 256;
      ro[r] = *(const ushort4*)&ts[(e >> 7) * 516 + (e & 127) * 4];
    }
#pragma unroll
    for (int r = 0; r < 16; ++r) {
      int e = tid + r * 256;
      *(ushort4*)&tbf[((size_t)(m0 + (e >> 7))) * 512 + (e & 127) * 4] = ro[r];
    }
  }
}

// ---------------------------------------------------------------------------
// Kernel 4: QKV GEMM  (65536x512)x(512x1536) bf16 WMMA, f32 acc.
// 128x128 block tile, 8 waves x (32x64). Register-staged, software-pipelined
// global->LDS copies. Output transposed via LDS to qkv[(sel,b,hd)][n] bf16.
// ---------------------------------------------------------------------------
__global__ void __launch_bounds__(256, 2)
qkv_gemm_kernel(const unsigned short* __restrict__ tbf,
                const unsigned short* __restrict__ wbf,
                unsigned short* __restrict__ qkv) {
  __shared__ unsigned short As[128 * 40];   // [m][k], pad to 40
  __shared__ unsigned short Bs[32 * 136];   // [k][j], pad to 136
  __shared__ unsigned short Ct[128 * 136];  // [j][m], pad to 136
  const int tid = threadIdx.x, lane = tid & 31, wid = tid >> 5;
  const int wm = wid & 3, wn = wid >> 2;
  const int ln15 = lane & 15, hi8 = (lane & 16) ? 8 : 0, kk = hi8;
  const int m0 = blockIdx.x * 128;
  const int b = m0 >> 12, n0 = m0 & 4095;
  const int jb = blockIdx.y * 128;
  v8f acc[2][4];
#pragma unroll
  for (int i = 0; i < 2; ++i)
#pragma unroll
    for (int j = 0; j < 4; ++j) acc[i][j] = {};
  const unsigned short* Abase = tbf + (size_t)m0 * 512;
  ushort4 ra[4], rb[4];
  auto loadTiles = [&](int kt) {
    const int k0 = kt * 32;
#pragma unroll
    for (int r = 0; r < 4; ++r) {
      int e = tid + r * 256;
      ra[r] = *(const ushort4*)(Abase + (size_t)(e >> 3) * 512 + k0 + (e & 7) * 4);
    }
#pragma unroll
    for (int r = 0; r < 4; ++r) {
      int e = tid + r * 256;
      rb[r] = *(const ushort4*)(wbf + (size_t)(k0 + (e >> 5)) * 1536 + jb + (e & 31) * 4);
    }
  };
  loadTiles(0);
  for (int kt = 0; kt < 16; ++kt) {
#pragma unroll
    for (int r = 0; r < 4; ++r) {
      int e = tid + r * 256;
      *(ushort4*)&As[(e >> 3) * 40 + (e & 7) * 4] = ra[r];
    }
#pragma unroll
    for (int r = 0; r < 4; ++r) {
      int e = tid + r * 256;
      *(ushort4*)&Bs[(e >> 5) * 136 + (e & 31) * 4] = rb[r];
    }
    __syncthreads();
    if (kt + 1 < 16) loadTiles(kt + 1);   // overlap next tile with compute
    BF16Frag a[2], bf[4];
#pragma unroll
    for (int i = 0; i < 2; ++i) {
      const __bf16* p = (const __bf16*)&As[(wm * 32 + i * 16 + ln15) * 40 + kk];
      a[i].h[0] = *(const v8bf*)p;
      a[i].h[1] = *(const v8bf*)(p + 16);
    }
#pragma unroll
    for (int j = 0; j < 4; ++j) {
      const __bf16* p = (const __bf16*)&Bs[kk * 136 + wn * 64 + j * 16 + ln15];
#pragma unroll
      for (int s = 0; s < 8; ++s) { bf[j].v[s] = p[s * 136]; bf[j].v[8 + s] = p[(16 + s) * 136]; }
    }
#pragma unroll
    for (int i = 0; i < 2; ++i)
#pragma unroll
      for (int j = 0; j < 4; ++j)
        acc[i][j] = __builtin_amdgcn_wmma_f32_16x16x32_bf16(
            false, a[i].v, false, bf[j].v, (short)0, acc[i][j], false, false);
    __syncthreads();
  }
  // epilogue: transpose through LDS (single 128-column pass)
#pragma unroll
  for (int i = 0; i < 2; ++i)
#pragma unroll
    for (int j = 0; j < 4; ++j)
#pragma unroll
      for (int r = 0; r < 8; ++r) {
        int m = wm * 32 + i * 16 + r + hi8;
        int jl = wn * 64 + j * 16 + ln15;
        Ct[jl * 136 + m] = f2bf(acc[i][j][r]);
      }
  __syncthreads();
  {
    ushort4 ro[16];
#pragma unroll
    for (int r = 0; r < 16; ++r) {
      int e = tid + r * 256;
      ro[r] = *(const ushort4*)&Ct[(e >> 5) * 136 + (e & 31) * 4];
    }
#pragma unroll
    for (int r = 0; r < 16; ++r) {
      int e = tid + r * 256;
      int jg = jb + (e >> 5);
      int sel = jg >> 9, hd = jg & 511;
      *(ushort4*)(qkv + ((size_t)((sel * 16 + b) * 512 + hd)) * 4096 + n0 + (e & 31) * 4) = ro[r];
    }
  }
}

// ---------------------------------------------------------------------------
// Kernel 5: reciprocal L2 norms of q,k rows (16384 rows x 4096), wave per row
// ---------------------------------------------------------------------------
__global__ void norm_kernel(const unsigned short* __restrict__ qkv, float* __restrict__ rn) {
  const int tid = threadIdx.x, lane = tid & 31, wid = tid >> 5;
  const int row = blockIdx.x * 8 + wid;
  const unsigned short* src = qkv + (size_t)row * 4096;
  float s = 0.f;
#pragma unroll 8
  for (int it = 0; it < 32; ++it) {
    ushort4 u = *(const ushort4*)(src + it * 128 + lane * 4);
    float a = bf2f(u.x), bq = bf2f(u.y), c = bf2f(u.z), d = bf2f(u.w);
    s += a * a + bq * bq + c * c + d * d;
  }
#pragma unroll
  for (int m = 16; m >= 1; m >>= 1) s += __shfl_xor(s, m, 32);
  if (lane == 0) rn[row] = 1.0f / fmaxf(sqrtf(s), 1e-12f);
}

// ---------------------------------------------------------------------------
// Kernel 6: per-(b,h) channel attention: gram(WMMA) -> softmax -> PV(WMMA)
// Normalization/temperature/scale folded into the score scaling.
// ---------------------------------------------------------------------------
__global__ void __launch_bounds__(256, 2)
attn_kernel(const unsigned short* __restrict__ qkv,
            const float* __restrict__ rn,
            const float* __restrict__ temp,
            unsigned short* __restrict__ ao) {
  __shared__ __align__(16) char smem[62976];
  unsigned short* Qs = (unsigned short*)smem;              // 64 x 136
  unsigned short* Kt = (unsigned short*)(smem + 17408);    // 128 x 72 (transposed)
  float*          S  = (float*)smem;                       // 64 x 68 (reuses Qs)
  unsigned short* P  = (unsigned short*)(smem + 17408);    // 64 x 72 (reuses Kt)
  unsigned short* Vs = (unsigned short*)(smem + 26624);    // 64 x 136
  unsigned short* Ot = (unsigned short*)(smem + 44032);    // 128 x 72
  float* srq = (float*)(smem + 62464);
  float* srk = (float*)(smem + 62720);
  const int tid = threadIdx.x, lane = tid & 31, wid = tid >> 5;
  const int ln15 = lane & 15, hi8 = (lane & 16) ? 8 : 0, kk = hi8;
  const int b = blockIdx.x >> 3, h = blockIdx.x & 7;
  const unsigned short* qp = qkv + ((size_t)((0 * 16 + b) * 512 + h * 64)) * 4096;
  const unsigned short* kp = qkv + ((size_t)((1 * 16 + b) * 512 + h * 64)) * 4096;
  const unsigned short* vp = qkv + ((size_t)((2 * 16 + b) * 512 + h * 64)) * 4096;
  if (tid < 64) srq[tid] = rn[b * 512 + h * 64 + tid] * expf(temp[h]) * 8.0f;
  else if (tid < 128) { int j = tid - 64; srk[j] = rn[8192 + b * 512 + h * 64 + j]; }
  // --- gram: G[64][64] = Q K^T over n, pipelined over 128-col chunks ---
  const int mi = (wid & 3) * 16, cj = (wid >> 2) * 32;
  v8f g0 = {}, g1 = {};
  ushort4 rq[8], rk[8];
  auto loadQK = [&](int it) {
    const int nb = it * 128;
#pragma unroll
    for (int r = 0; r < 8; ++r) {
      int e = tid + r * 256;
      int d = e >> 5, seg = e & 31;
      rq[r] = *(const ushort4*)(qp + (size_t)d * 4096 + nb + seg * 4);
      rk[r] = *(const ushort4*)(kp + (size_t)d * 4096 + nb + seg * 4);
    }
  };
  loadQK(0);
  for (int it = 0; it < 32; ++it) {
#pragma unroll
    for (int r = 0; r < 8; ++r) {
      int e = tid + r * 256;
      int d = e >> 5, seg = e & 31, nl = seg * 4;
      *(ushort4*)&Qs[d * 136 + seg * 4] = rq[r];
      Kt[(nl + 0) * 72 + d] = rk[r].x; Kt[(nl + 1) * 72 + d] = rk[r].y;
      Kt[(nl + 2) * 72 + d] = rk[r].z; Kt[(nl + 3) * 72 + d] = rk[r].w;
    }
    __syncthreads();
    if (it + 1 < 32) loadQK(it + 1);
#pragma unroll
    for (int kf = 0; kf < 4; ++kf) {
      const int k0 = kf * 32;
      BF16Frag a, b0, b1;
      const __bf16* ap_ = (const __bf16*)&Qs[(mi + ln15) * 136 + k0 + kk];
      a.h[0] = *(const v8bf*)ap_; a.h[1] = *(const v8bf*)(ap_ + 16);
      const __bf16* bp_ = (const __bf16*)&Kt[(k0 + kk) * 72 + cj + ln15];
#pragma unroll
      for (int s = 0; s < 8; ++s) {
        b0.v[s] = bp_[s * 72]; b0.v[8 + s] = bp_[(16 + s) * 72];
        b1.v[s] = bp_[s * 72 + 16]; b1.v[8 + s] = bp_[(16 + s) * 72 + 16];
      }
      g0 = __builtin_amdgcn_wmma_f32_16x16x32_bf16(false, a.v, false, b0.v, (short)0, g0, false, false);
      g1 = __builtin_amdgcn_wmma_f32_16x16x32_bf16(false, a.v, false, b1.v, (short)0, g1, false, false);
    }
    __syncthreads();
  }
  // --- scores with folded scale ---
#pragma unroll
  for (int r = 0; r < 8; ++r) {
    int i = mi + r + hi8;
    S[i * 68 + cj + ln15]      = g0[r] * srq[i] * srk[cj + ln15];
    S[i * 68 + cj + 16 + ln15] = g1[r] * srq[i] * srk[cj + 16 + ln15];
  }
  __syncthreads();
  if (tid < 64) {
    float mx = -1e30f;
#pragma unroll 8
    for (int j = 0; j < 64; ++j) mx = fmaxf(mx, S[tid * 68 + j]);
    float sum = 0.f;
#pragma unroll 8
    for (int j = 0; j < 64; ++j) sum += expf(S[tid * 68 + j] - mx);
    float inv = 1.0f / sum;
#pragma unroll 8
    for (int j = 0; j < 64; ++j) P[tid * 72 + j] = f2bf(expf(S[tid * 68 + j] - mx) * inv);
  }
  __syncthreads();
  // --- PV: out[64][n] = P @ V, streamed over n, stored token-major bf16 ---
  const int mi2 = (wid & 3) * 16, co = (wid >> 2) * 64;
  BF16Frag ap[2];
#pragma unroll
  for (int kf = 0; kf < 2; ++kf) {
    const __bf16* p = (const __bf16*)&P[(mi2 + ln15) * 72 + kf * 32 + kk];
    ap[kf].h[0] = *(const v8bf*)p; ap[kf].h[1] = *(const v8bf*)(p + 16);
  }
  ushort4 rv[8];
  auto loadV = [&](int it) {
    const int nb = it * 128;
#pragma unroll
    for (int r = 0; r < 8; ++r) {
      int e = tid + r * 256;
      rv[r] = *(const ushort4*)(vp + (size_t)(e >> 5) * 4096 + nb + (e & 31) * 4);
    }
  };
  loadV(0);
  for (int it = 0; it < 32; ++it) {
    const int nb = it * 128;
#pragma unroll
    for (int r = 0; r < 8; ++r) {
      int e = tid + r * 256;
      *(ushort4*)&Vs[(e >> 5) * 136 + (e & 31) * 4] = rv[r];
    }
    __syncthreads();
    if (it + 1 < 32) loadV(it + 1);
    v8f oacc[4];
#pragma unroll
    for (int t = 0; t < 4; ++t) oacc[t] = {};
#pragma unroll
    for (int kf = 0; kf < 2; ++kf) {
      const int k0 = kf * 32;
#pragma unroll
      for (int t = 0; t < 4; ++t) {
        BF16Frag bv;
        const __bf16* p = (const __bf16*)&Vs[(k0 + kk) * 136 + co + t * 16 + ln15];
#pragma unroll
        for (int s = 0; s < 8; ++s) { bv.v[s] = p[s * 136]; bv.v[8 + s] = p[(16 + s) * 136]; }
        oacc[t] = __builtin_amdgcn_wmma_f32_16x16x32_bf16(
            false, ap[kf].v, false, bv.v, (short)0, oacc[t], false, false);
      }
    }
#pragma unroll
    for (int t = 0; t < 4; ++t)
#pragma unroll
      for (int r = 0; r < 8; ++r) {
        int d = mi2 + r + hi8;
        int nl = co + t * 16 + ln15;
        Ot[nl * 72 + d] = f2bf(oacc[t][r]);
      }
    __syncthreads();
    {
      ushort4 ro[8];
#pragma unroll
      for (int r = 0; r < 8; ++r) {
        int e = tid + r * 256;
        ro[r] = *(const ushort4*)&Ot[(e >> 4) * 72 + (e & 15) * 4];
      }
#pragma unroll
      for (int r = 0; r < 8; ++r) {
        int e = tid + r * 256;
        *(ushort4*)(ao + ((size_t)(b * 4096 + nb + (e >> 4))) * 512 + h * 64 + (e & 15) * 4) = ro[r];
      }
    }
    __syncthreads();
  }
}

// ---------------------------------------------------------------------------
// Kernel 7: output GEMM (65536x512)x(512x512) + bias, f32 result transposed
// straight into d_out's (b,c,h,w) layout. 128x128 block, 8 waves x (32x64).
// ---------------------------------------------------------------------------
__global__ void __launch_bounds__(256, 2)
out_gemm_kernel(const unsigned short* __restrict__ abf,
                const unsigned short* __restrict__ wbf,
                const float* __restrict__ bias,
                float* __restrict__ out) {
  __shared__ unsigned short As[128 * 40];
  __shared__ unsigned short Bs[32 * 136];
  __shared__ float Ct[64 * 132];
  const int tid = threadIdx.x, lane = tid & 31, wid = tid >> 5;
  const int wm = wid & 3, wn = wid >> 2;
  const int ln15 = lane & 15, hi8 = (lane & 16) ? 8 : 0, kk = hi8;
  const int m0 = blockIdx.x * 128;
  const int b = m0 >> 12, n0 = m0 & 4095;
  const int jb = blockIdx.y * 128;
  v8f acc[2][4];
#pragma unroll
  for (int i = 0; i < 2; ++i)
#pragma unroll
    for (int j = 0; j < 4; ++j) acc[i][j] = {};
  const unsigned short* Abase = abf + (size_t)m0 * 512;
  ushort4 ra[4], rb[4];
  auto loadTiles = [&](int kt) {
    const int k0 = kt * 32;
#pragma unroll
    for (int r = 0; r < 4; ++r) {
      int e = tid + r * 256;
      ra[r] = *(const ushort4*)(Abase + (size_t)(e >> 3) * 512 + k0 + (e & 7) * 4);
    }
#pragma unroll
    for (int r = 0; r < 4; ++r) {
      int e = tid + r * 256;
      rb[r] = *(const ushort4*)(wbf + (size_t)(k0 + (e >> 5)) * 512 + jb + (e & 31) * 4);
    }
  };
  loadTiles(0);
  for (int kt = 0; kt < 16; ++kt) {
#pragma unroll
    for (int r = 0; r < 4; ++r) {
      int e = tid + r * 256;
      *(ushort4*)&As[(e >> 3) * 40 + (e & 7) * 4] = ra[r];
    }
#pragma unroll
    for (int r = 0; r < 4; ++r) {
      int e = tid + r * 256;
      *(ushort4*)&Bs[(e >> 5) * 136 + (e & 31) * 4] = rb[r];
    }
    __syncthreads();
    if (kt + 1 < 16) loadTiles(kt + 1);
    BF16Frag a[2], bf[4];
#pragma unroll
    for (int i = 0; i < 2; ++i) {
      const __bf16* p = (const __bf16*)&As[(wm * 32 + i * 16 + ln15) * 40 + kk];
      a[i].h[0] = *(const v8bf*)p;
      a[i].h[1] = *(const v8bf*)(p + 16);
    }
#pragma unroll
    for (int j = 0; j < 4; ++j) {
      const __bf16* p = (const __bf16*)&Bs[kk * 136 + wn * 64 + j * 16 + ln15];
#pragma unroll
      for (int s = 0; s < 8; ++s) { bf[j].v[s] = p[s * 136]; bf[j].v[8 + s] = p[(16 + s) * 136]; }
    }
#pragma unroll
    for (int i = 0; i < 2; ++i)
#pragma unroll
      for (int j = 0; j < 4; ++j)
        acc[i][j] = __builtin_amdgcn_wmma_f32_16x16x32_bf16(
            false, a[i].v, false, bf[j].v, (short)0, acc[i][j], false, false);
    __syncthreads();
  }
  // epilogue: two passes of 64 columns, f32 transpose + bias
  for (int p = 0; p < 2; ++p) {
    if (wn == p) {
#pragma unroll
      for (int i = 0; i < 2; ++i)
#pragma unroll
        for (int j = 0; j < 4; ++j)
#pragma unroll
          for (int r = 0; r < 8; ++r) {
            int m = wm * 32 + i * 16 + r + hi8;
            int jl = j * 16 + ln15;
            Ct[jl * 132 + m] = acc[i][j][r];
          }
    }
    __syncthreads();
    {
      float4 ro[8];
#pragma unroll
      for (int r = 0; r < 8; ++r) {
        int e = tid + r * 256;
        ro[r] = *(const float4*)&Ct[(e >> 5) * 132 + (e & 31) * 4];
      }
#pragma unroll
      for (int r = 0; r < 8; ++r) {
        int e = tid + r * 256;
        int jg = jb + p * 64 + (e >> 5);
        float bv = bias[jg];
        float4 v = ro[r];
        v.x += bv; v.y += bv; v.z += bv; v.w += bv;
        *(float4*)(out + ((size_t)(b * 512 + jg)) * 4096 + n0 + (e & 31) * 4) = v;
      }
    }
    __syncthreads();
  }
}

// ---------------------------------------------------------------------------
extern "C" void kernel_launch(void* const* d_in, const int* in_sizes, int n_in,
                              void* d_out, int out_size, void* d_ws, size_t ws_size,
                              hipStream_t stream) {
  (void)in_sizes; (void)n_in; (void)out_size; (void)ws_size;
  const float* x    = (const float*)d_in[0];
  const float* cond = (const float*)d_in[1];
  const float* fw1  = (const float*)d_in[2];
  const float* fb1  = (const float*)d_in[3];
  const float* fw2  = (const float*)d_in[4];
  const float* fb2  = (const float*)d_in[5];
  const float* qkvw = (const float*)d_in[6];
  const float* temp = (const float*)d_in[7];
  const float* outw = (const float*)d_in[8];
  const float* outb = (const float*)d_in[9];
  float* out = (float*)d_out;
  char* ws = (char*)d_ws;
  // workspace layout (bytes)
  float* gb            = (float*)(ws + 0);              //  16*1024 f32
  float* rn            = (float*)(ws + 65536);          //  2*16*512 f32
  unsigned short* qwbf = (unsigned short*)(ws + 131072);    // 512*1536 bf16
  unsigned short* owbf = (unsigned short*)(ws + 1703936);   // 512*512 bf16
  unsigned short* tbf  = (unsigned short*)(ws + 2228224);   // 65536*512 bf16
  unsigned short* qkvb = (unsigned short*)(ws + 69337088);  // 3*16*512*4096 bf16
  unsigned short* aob  = (unsigned short*)(ws + 270663680); // 65536*512 bf16

  hipLaunchKernelGGL(film_kernel,    dim3(16),       dim3(256), 0, stream, cond, fw1, fb1, fw2, fb2, gb);
  hipLaunchKernelGGL(cvtw_kernel,    dim3(4096),     dim3(256), 0, stream, qkvw, outw, qwbf, owbf);
  hipLaunchKernelGGL(ln_kernel,      dim3(2048),     dim3(256), 0, stream, x, gb, tbf);
  hipLaunchKernelGGL(qkv_gemm_kernel,dim3(512, 12),  dim3(256), 0, stream, tbf, qwbf, qkvb);
  hipLaunchKernelGGL(norm_kernel,    dim3(2048),     dim3(256), 0, stream, qkvb, rn);
  hipLaunchKernelGGL(attn_kernel,    dim3(128),      dim3(256), 0, stream, qkvb, rn, temp, aob);
  hipLaunchKernelGGL(out_gemm_kernel,dim3(512, 4),   dim3(256), 0, stream, aob, owbf, outb, out);
}
